// Synapformer_mamba_59923383714432
// MI455X (gfx1250) — matrix operand order repaired and checked by reading, not compiled, not run on hardware
//
#include <hip/hip_runtime.h>
#include <hip/hip_bf16.h>
#include <math.h>

#define N_NODES 8192
#define F_IN    128
#define HID     64
#define HEADS   8
#define REP     512
#define NCLS    32
#define E_EDGES 2048
#define NSYN    4096
#define LMAX    8
#define DM      512
#define DI      1024
#define DSTATE  16
#define DCONV   4
#define DTR     32
#define EDGES_TOT (E_EDGES + N_NODES)   // 10240 (graph edges + self loops)
#define SEQ_TOT   (E_EDGES * LMAX)      // 16384 padded sequence rows

typedef __attribute__((ext_vector_type(16))) __bf16 v16bf;
typedef __attribute__((ext_vector_type(8)))  float  v8f;

// ---------------------------------------------------------------- helpers
__device__ __forceinline__ __bf16 f2bf(float f) {
  unsigned u = __builtin_bit_cast(unsigned, f);
  unsigned r = u + 0x7FFFu + ((u >> 16) & 1u);          // round-to-nearest-even
  return __builtin_bit_cast(__bf16, (unsigned short)(r >> 16));
}
__device__ __forceinline__ unsigned fkey(float f) {     // order-preserving uint key
  unsigned u = __builtin_bit_cast(unsigned, f);
  return (u & 0x80000000u) ? ~u : (u | 0x80000000u);
}
__device__ __forceinline__ float funkey(unsigned k) {
  unsigned u = (k & 0x80000000u) ? (k & 0x7FFFFFFFu) : ~k;
  return __builtin_bit_cast(float, u);
}
__device__ __forceinline__ float siluf(float x) { return x / (1.f + expf(-x)); }

// ---------------------------------------------------------------- WMMA GEMM
// C[M,N] = act(A[M,K] @ B[K,N] + bias), f32 in/out, bf16 multiply, f32 accum.
// 64x64 tile, 128 threads = 4 wave32 waves, each wave a 2x2 grid of
// v_wmma_f32_16x16x32_bf16 accumulators. Double-buffered LDS, software
// pipeline: loads(t+1) || wmma(t) ; convert+store(t+1) ; barrier.
//
// LDS tiles are stored in WMMA *fragment order* so each lane's 16-bf16
// fragment is one aligned 32-byte contiguous chunk:
//   AsF[((m*2)+half)*16 + e]  with  k = e + 16*half
//   BsF[n*32 + k]             with  k = e + 16*half
// act: 0=none 1=relu 2=softplus

// load phase: issue all global loads for one 64x64/32 tile into registers
__device__ __forceinline__ void stage_load(const float* __restrict__ A, int lda,
                                           const float* __restrict__ B, int ldb,
                                           int tileM, int tileN, int k0, int K, int tid,
                                           float4 aR[4], float bR[4][4])
{
#pragma unroll
  for (int it = 0; it < 4; ++it) {
    const int idx = tid + 128 * it;            // 512 float4 chunks of A (64x32)
    const int r  = idx >> 3;
    const int kq = (idx & 7) * 4;
    aR[it] = *(const float4*)(A + (size_t)(tileM + r) * lda + k0 + kq);
  }
#pragma unroll
  for (int it = 0; it < 4; ++it) {
    const int idx = tid + 128 * it;            // 512 (n, k-quad) pairs of B (32x64)
    const int n  = idx & 63;
    const int kq = (idx >> 6) * 4;
#pragma unroll
    for (int j = 0; j < 4; ++j)
      bR[it][j] = B[(size_t)(k0 + kq + j) * ldb + tileN + n];
  }
  // speculative prefetch one tile further ahead (global_prefetch_b8)
  if (k0 + 32 < K) {
    __builtin_prefetch(A + (size_t)(tileM + (tid & 63)) * lda + (k0 + 32), 0, 1);
    __builtin_prefetch(B + (size_t)(k0 + 32 + (tid & 31)) * ldb + tileN, 0, 1);
  }
}

// store phase: convert registers to bf16 fragment order in LDS
__device__ __forceinline__ void stage_store(__bf16* AsF, __bf16* BsF, int tid,
                                            const float4 aR[4], const float bR[4][4])
{
#pragma unroll
  for (int it = 0; it < 4; ++it) {
    const int idx = tid + 128 * it;
    const int r  = idx >> 3;
    const int kq = (idx & 7) * 4;
    const int h  = kq >> 4;
    const int eb = kq & 15;
    union { __bf16 b[4]; uint2 u; } pk;
    pk.b[0] = f2bf(aR[it].x); pk.b[1] = f2bf(aR[it].y);
    pk.b[2] = f2bf(aR[it].z); pk.b[3] = f2bf(aR[it].w);
    *(uint2*)&AsF[(r * 2 + h) * 16 + eb] = pk.u;     // ds_store_b64
  }
#pragma unroll
  for (int it = 0; it < 4; ++it) {
    const int idx = tid + 128 * it;
    const int n  = idx & 63;
    const int kq = (idx >> 6) * 4;
    union { __bf16 b[4]; uint2 u; } pk;
#pragma unroll
    for (int j = 0; j < 4; ++j) pk.b[j] = f2bf(bR[it][j]);
    *(uint2*)&BsF[n * 32 + kq] = pk.u;               // ds_store_b64
  }
}

// fragments: one aligned 32B LDS read each (2x ds_load_b128), then 4 WMMA
__device__ __forceinline__ void tile_mma(const __bf16* AsF, const __bf16* BsF,
                                         int wm, int wn, int l16, int half,
                                         v8f acc[2][2])
{
  v16bf af[2], bfm[2];
#pragma unroll
  for (int mi = 0; mi < 2; ++mi)
    af[mi] = *(const v16bf*)&AsF[((wm + mi * 16 + l16) * 2 + half) * 16];
#pragma unroll
  for (int ni = 0; ni < 2; ++ni)
    bfm[ni] = *(const v16bf*)&BsF[(wn + ni * 16 + l16) * 32 + 16 * half];
#pragma unroll
  for (int mi = 0; mi < 2; ++mi)
#pragma unroll
    for (int ni = 0; ni < 2; ++ni)
      acc[mi][ni] = __builtin_amdgcn_wmma_f32_16x16x32_bf16(
          false, af[mi], false, bfm[ni], (short)0, acc[mi][ni], false, false);
}

template <bool FULL>
__global__ __launch_bounds__(128)
void wmma_gemm_bf16(const float* __restrict__ A, int lda,
                    const float* __restrict__ B, int ldb,
                    float* __restrict__ C, int ldc,
                    int M, int N, int K,
                    const float* __restrict__ bias, int act)
{
  __shared__ __bf16 AsF[2][64 * 32];   // double-buffered fragment-ordered tiles
  __shared__ __bf16 BsF[2][64 * 32];

  const int tid  = threadIdx.x;
  const int lane = tid & 31;
  const int wave = tid >> 5;
  const int wm   = (wave >> 1) * 32;        // wave sub-tile origin inside 64x64
  const int wn   = (wave & 1) * 32;
  const int half = lane >> 4;
  const int l16  = lane & 15;
  const int tileM = blockIdx.y * 64;
  const int tileN = blockIdx.x * 64;

  v8f zero = {};
  v8f acc[2][2];
  acc[0][0] = zero; acc[0][1] = zero; acc[1][0] = zero; acc[1][1] = zero;

  if (FULL) {
    const int nt = K >> 5;                  // K % 32 == 0
    float4 aR[4]; float bR[4][4];
    stage_load(A, lda, B, ldb, tileM, tileN, 0, K, tid, aR, bR);
    stage_store(AsF[0], BsF[0], tid, aR, bR);
    __syncthreads();
    for (int t = 0; t < nt; ++t) {
      const int cur = t & 1;
      const bool more = (t + 1 < nt);
      if (more) stage_load(A, lda, B, ldb, tileM, tileN, (t + 1) * 32, K, tid, aR, bR);
      tile_mma(AsF[cur], BsF[cur], wm, wn, l16, half, acc);       // hides load latency
      if (more) stage_store(AsF[cur ^ 1], BsF[cur ^ 1], tid, aR, bR);
      __syncthreads();
    }
  } else {
    // ---- generic guarded path (ragged tiles: K=6 input proj, N=32 head)
    for (int k0 = 0; k0 < K; k0 += 32) {
      for (int i = tid; i < 64 * 32; i += 128) {
        const int r = i >> 5, k = i & 31;
        const int gr = tileM + r, gc = k0 + k;
        const float v = (gr < M && gc < K) ? A[(size_t)gr * lda + gc] : 0.f;
        const int h = k >> 4;
        const int e = k & 15;
        AsF[0][(r * 2 + h) * 16 + e] = f2bf(v);
      }
      for (int i = tid; i < 32 * 64; i += 128) {
        const int k = i >> 6, n = i & 63;
        const int gr = k0 + k, gc = tileN + n;
        const float v = (gr < K && gc < N) ? B[(size_t)gr * ldb + gc] : 0.f;
        BsF[0][n * 32 + k] = f2bf(v);
      }
      __syncthreads();
      tile_mma(AsF[0], BsF[0], wm, wn, l16, half, acc);
      __syncthreads();
    }
  }

  // C/D layout: vgpr r -> row r + 8*half, col = lane&15
#pragma unroll
  for (int mi = 0; mi < 2; ++mi)
#pragma unroll
    for (int ni = 0; ni < 2; ++ni)
#pragma unroll
      for (int r = 0; r < 8; ++r) {
        const int row = tileM + wm + mi * 16 + half * 8 + r;
        const int col = tileN + wn + ni * 16 + l16;
        if (FULL || (row < M && col < N)) {
          float v = acc[mi][ni][r];
          if (bias) v += bias[col];
          if (act == 1)      v = fmaxf(v, 0.f);
          else if (act == 2) v = (v > 20.f) ? v : log1pf(expf(v)); // softplus
          C[(size_t)row * ldc + col] = v;
        }
      }
}

// ---------------------------------------------------------------- GAT kernels
__global__ void attn_dots(const float* __restrict__ xw,
                          const float* __restrict__ a_s,
                          const float* __restrict__ a_d,
                          float* __restrict__ out_s, float* __restrict__ out_d,
                          int H, int C)
{
  int t = blockIdx.x * blockDim.x + threadIdx.x;
  if (t >= N_NODES * H) return;
  int node = t / H, h = t % H;
  const float* x = xw + ((size_t)node * H + h) * C;
  float ss = 0.f, dd = 0.f;
  for (int c = 0; c < C; ++c) { float v = x[c]; ss += v * a_s[h * C + c]; dd += v * a_d[h * C + c]; }
  out_s[t] = ss; out_d[t] = dd;
}

__global__ void edge_scores(const float* __restrict__ asrc, const float* __restrict__ adst,
                            const int* __restrict__ ei,
                            float* __restrict__ esc, unsigned* __restrict__ mkey, int H)
{
  int t = blockIdx.x * blockDim.x + threadIdx.x;
  if (t >= EDGES_TOT * H) return;
  int e = t / H, h = t % H;
  int s = (e < E_EDGES) ? ei[e]           : e - E_EDGES;
  int d = (e < E_EDGES) ? ei[E_EDGES + e] : e - E_EDGES;
  float v = asrc[s * H + h] + adst[d * H + h];
  v = (v > 0.f) ? v : 0.2f * v;                       // leaky_relu(0.2)
  esc[t] = v;
  atomicMax(&mkey[d * H + h], fkey(v));
}

__global__ void edge_exp(float* __restrict__ esc, const unsigned* __restrict__ mkey,
                         float* __restrict__ den, const int* __restrict__ ei, int H)
{
  int t = blockIdx.x * blockDim.x + threadIdx.x;
  if (t >= EDGES_TOT * H) return;
  int e = t / H, h = t % H;
  int d = (e < E_EDGES) ? ei[E_EDGES + e] : e - E_EDGES;
  float ex = expf(esc[t] - funkey(mkey[d * H + h]));
  esc[t] = ex;
  atomicAdd(&den[d * H + h], ex);
}

__global__ void edge_agg(const float* __restrict__ xw, const float* __restrict__ ex,
                         const float* __restrict__ den, const int* __restrict__ ei,
                         float* __restrict__ agg, int H, int C)
{
  long long t = (long long)blockIdx.x * blockDim.x + threadIdx.x;
  long long tot = (long long)EDGES_TOT * H * C;
  if (t >= tot) return;
  int c = (int)(t % C); int h = (int)((t / C) % H); int e = (int)(t / ((long long)C * H));
  int s = (e < E_EDGES) ? ei[e]           : e - E_EDGES;
  int d = (e < E_EDGES) ? ei[E_EDGES + e] : e - E_EDGES;
  float alpha = ex[e * H + h] / den[d * H + h];
  atomicAdd(&agg[((size_t)d * H + h) * C + c], alpha * xw[((size_t)s * H + h) * C + c]);
}

// act: 0 = none, 3 = elu
__global__ void bias_act(float* __restrict__ y, const float* __restrict__ x,
                         const float* __restrict__ b, long long total, int cols, int act)
{
  long long t = (long long)blockIdx.x * blockDim.x + threadIdx.x;
  if (t >= total) return;
  float v = x[t] + b[(int)(t % cols)];
  if (act == 3) v = (v > 0.f) ? v : (expf(v) - 1.f);
  y[t] = v;
}

// ---------------------------------------------------------------- Mamba kernels
__global__ void seq_counts(const int* __restrict__ sidx, int* __restrict__ counts)
{
  int t = blockIdx.x * blockDim.x + threadIdx.x;
  if (t >= NSYN) return;
  atomicAdd(&counts[sidx[t]], 1);
}
__global__ void seq_starts(const int* __restrict__ counts, int* __restrict__ starts)
{
  if (blockIdx.x == 0 && threadIdx.x == 0) {
    int acc = 0;
    for (int e = 0; e < E_EDGES; ++e) { starts[e] = acc; acc += counts[e]; }
  }
}
__global__ void seq_scatter(const float* __restrict__ synapse, const int* __restrict__ sidx,
                            const int* __restrict__ starts, float* __restrict__ padded)
{
  int t = blockIdx.x * blockDim.x + threadIdx.x;
  if (t >= NSYN) return;
  int idx = sidx[t];
  int pos = t - starts[idx];
  if (pos >= LMAX) return;                         // mode='drop'
  for (int c = 0; c < 6; ++c)
    padded[((size_t)idx * LMAX + pos) * 6 + c] = synapse[(size_t)t * 6 + c];
}

// depthwise causal conv(k=4) + silu;  xm = xz[:, :DI]
__global__ void dwconv_silu(const float* __restrict__ xz, const float* __restrict__ w,
                            const float* __restrict__ b, float* __restrict__ xc)
{
  long long t = (long long)blockIdx.x * blockDim.x + threadIdx.x;
  if (t >= (long long)SEQ_TOT * DI) return;
  int d = (int)(t % DI);
  int l = (int)((t / DI) % LMAX);
  int e = (int)(t / ((long long)DI * LMAX));
  float acc = b[d];
#pragma unroll
  for (int k = 0; k < DCONV; ++k) {
    int ll = l + k - (DCONV - 1);
    if (ll >= 0) acc += w[d * DCONV + k] * xz[((size_t)e * LMAX + ll) * (2 * DI) + d];
  }
  xc[t] = siluf(acc);
}

// selective-state scan: one thread per (edge, channel), 16 states
__global__ void ssm_scan(const float* __restrict__ dt, const float* __restrict__ xdb,
                         const float* __restrict__ xc, const float* __restrict__ A_log,
                         const float* __restrict__ Dv, float* __restrict__ y)
{
  int t = blockIdx.x * blockDim.x + threadIdx.x;
  if (t >= E_EDGES * DI) return;
  int d = t % DI, e = t / DI;
  float Av[DSTATE], h[DSTATE];
#pragma unroll
  for (int n = 0; n < DSTATE; ++n) { Av[n] = -expf(A_log[d * DSTATE + n]); h[n] = 0.f; }
  for (int l = 0; l < LMAX; ++l) {
    size_t base = (size_t)e * LMAX + l;
    float dtv = dt[base * DI + d];
    float xv  = xc[base * DI + d];
    const float* Bp = xdb + base * (DTR + 2 * DSTATE) + DTR;
    const float* Cp = Bp + DSTATE;
    float yv = 0.f;
#pragma unroll
    for (int n = 0; n < DSTATE; ++n) {
      h[n] = expf(dtv * Av[n]) * h[n] + dtv * xv * Bp[n];
      yv += h[n] * Cp[n];
    }
    y[base * DI + d] = yv + Dv[d] * xv;
  }
}

__global__ void gate_silu(float* __restrict__ y, const float* __restrict__ xz)
{
  long long t = (long long)blockIdx.x * blockDim.x + threadIdx.x;
  if (t >= (long long)SEQ_TOT * DI) return;
  int d = (int)(t % DI);
  long long base = t / DI;
  float z = xz[(size_t)base * (2 * DI) + DI + d];
  y[t] *= siluf(z);
}

__global__ void pool_mean(const float* __restrict__ outseq, const int* __restrict__ counts,
                          float* __restrict__ pooled)
{
  int t = blockIdx.x * blockDim.x + threadIdx.x;
  if (t >= E_EDGES * DM) return;
  int d = t % DM, e = t / DM;
  int c = counts[e]; if (c < 1) c = 1; if (c > LMAX) c = LMAX;
  float s = 0.f;
  for (int l = 0; l < c; ++l) s += outseq[((size_t)e * LMAX + l) * DM + d];
  pooled[t] = s / (float)c;
}

// ---------------------------------------------------------------- scatter-max
__global__ void seg_scatter(const float* __restrict__ x, const int* __restrict__ idx,
                            unsigned* __restrict__ key, int* __restrict__ nz)
{
  int t = blockIdx.x * blockDim.x + threadIdx.x;
  if (t >= E_EDGES * DM) return;
  int d = t % DM, e = t / DM;
  int node = idx[e];
  atomicMax(&key[(size_t)node * DM + d], fkey(x[t]));
  if (d == 0) atomicAdd(&nz[node], 1);
}
__global__ void seg_final(const unsigned* __restrict__ key, const int* __restrict__ nz,
                          float* __restrict__ out)
{
  long long t = (long long)blockIdx.x * blockDim.x + threadIdx.x;
  if (t >= (long long)N_NODES * DM) return;
  int n = (int)(t / DM);
  out[t] = (nz[n] > 0) ? funkey(key[t]) : 0.f;
}

__global__ void build_feats(const float* __restrict__ h2, const float* __restrict__ left,
                            const float* __restrict__ right, float* __restrict__ feats)
{
  long long t = (long long)blockIdx.x * blockDim.x + threadIdx.x;
  long long tot = (long long)N_NODES * (REP + 2 * DM);
  if (t >= tot) return;
  int n = (int)(t / (REP + 2 * DM));
  int c = (int)(t % (REP + 2 * DM));
  float v;
  if (c < REP)            v = h2[(size_t)n * REP + c];
  else if (c < REP + DM)  v = left[(size_t)n * DM + (c - REP)];
  else                    v = right[(size_t)n * DM + (c - REP - DM)];
  feats[t] = v;
}

// ---------------------------------------------------------------- launch
extern "C" void kernel_launch(void* const* d_in, const int* in_sizes, int n_in,
                              void* d_out, int out_size, void* d_ws, size_t ws_size,
                              hipStream_t stream)
{
  (void)in_sizes; (void)n_in; (void)out_size; (void)ws_size;
  const float* synapse   = (const float*)d_in[1];
  const float* x_param   = (const float*)d_in[2];
  const float* W1        = (const float*)d_in[3];
  const float* att_src1  = (const float*)d_in[4];
  const float* att_dst1  = (const float*)d_in[5];
  const float* b1        = (const float*)d_in[6];
  const float* W2        = (const float*)d_in[7];
  const float* att_src2  = (const float*)d_in[8];
  const float* att_dst2  = (const float*)d_in[9];
  const float* b2        = (const float*)d_in[10];
  const float* W_in      = (const float*)d_in[11];
  const float* b_in      = (const float*)d_in[12];
  const float* W_inproj  = (const float*)d_in[13];
  const float* conv_w    = (const float*)d_in[14];
  const float* conv_b    = (const float*)d_in[15];
  const float* W_xproj   = (const float*)d_in[16];
  const float* W_dt      = (const float*)d_in[17];
  const float* b_dt      = (const float*)d_in[18];
  const float* A_log     = (const float*)d_in[19];
  const float* Dvec      = (const float*)d_in[20];
  const float* W_outproj = (const float*)d_in[21];
  const float* W_lout    = (const float*)d_in[22];
  const float* b_lout    = (const float*)d_in[23];
  const float* Wc1       = (const float*)d_in[24];
  const float* bc1       = (const float*)d_in[25];
  const float* Wc2       = (const float*)d_in[26];
  const float* bc2       = (const float*)d_in[27];
  const int*   edge_index    = (const int*)d_in[28];
  const int*   synapse_index = (const int*)d_in[29];

  char* ws = (char*)d_ws;
  size_t off = 0;
  auto alloc = [&](size_t elems32) -> void* {
    void* p = ws + off;
    off += ((elems32 * 4 + 255) / 256) * 256;
    return p;
  };
  auto cdiv  = [](long long a, long long b) { return (unsigned)((a + b - 1) / b); };
  auto zero  = [&](void* p, size_t elems32) { hipMemsetAsync(p, 0, elems32 * 4, stream); };
  auto gemm  = [&](const float* A, int lda, const float* B, int ldb, float* C, int ldc,
                   int M, int N, int K, const float* bias, int act) {
    dim3 g(cdiv(N, 64), cdiv(M, 64));
    const bool full = (M % 64 == 0) && (N % 64 == 0) && (K % 32 == 0) && (lda % 4 == 0);
    if (full)
      wmma_gemm_bf16<true><<<g, dim3(128), 0, stream>>>(A, lda, B, ldb, C, ldc, M, N, K, bias, act);
    else
      wmma_gemm_bf16<false><<<g, dim3(128), 0, stream>>>(A, lda, B, ldb, C, ldc, M, N, K, bias, act);
  };

  // ---- workspace
  float* xw1   = (float*)alloc((size_t)N_NODES * HEADS * HID);
  float* asrc1 = (float*)alloc((size_t)N_NODES * HEADS);
  float* adst1 = (float*)alloc((size_t)N_NODES * HEADS);
  float* esc1  = (float*)alloc((size_t)EDGES_TOT * HEADS);
  unsigned* mkey1 = (unsigned*)alloc((size_t)N_NODES * HEADS);
  float* den1  = (float*)alloc((size_t)N_NODES * HEADS);
  float* agg1  = (float*)alloc((size_t)N_NODES * HEADS * HID);
  float* h1    = (float*)alloc((size_t)N_NODES * HEADS * HID);
  float* xw2   = (float*)alloc((size_t)N_NODES * REP);
  float* asrc2 = (float*)alloc((size_t)N_NODES);
  float* adst2 = (float*)alloc((size_t)N_NODES);
  float* esc2  = (float*)alloc((size_t)EDGES_TOT);
  unsigned* mkey2 = (unsigned*)alloc((size_t)N_NODES);
  float* den2  = (float*)alloc((size_t)N_NODES);
  float* agg2  = (float*)alloc((size_t)N_NODES * REP);
  float* h2    = (float*)alloc((size_t)N_NODES * REP);
  int*   counts = (int*)alloc(E_EDGES);
  int*   starts = (int*)alloc(E_EDGES);
  float* padded = (float*)alloc((size_t)SEQ_TOT * 6);
  float* u      = (float*)alloc((size_t)SEQ_TOT * DM);
  float* xz     = (float*)alloc((size_t)SEQ_TOT * 2 * DI);
  float* xc     = (float*)alloc((size_t)SEQ_TOT * DI);
  float* xdb    = (float*)alloc((size_t)SEQ_TOT * (DTR + 2 * DSTATE));
  float* dtb    = (float*)alloc((size_t)SEQ_TOT * DI);
  float* yb     = (float*)alloc((size_t)SEQ_TOT * DI);
  float* outseq = (float*)alloc((size_t)SEQ_TOT * DM);
  float* pooled = (float*)alloc((size_t)E_EDGES * DM);
  float* xpoint = (float*)alloc((size_t)E_EDGES * DM);
  unsigned* keyL = (unsigned*)alloc((size_t)N_NODES * DM);
  unsigned* keyR = (unsigned*)alloc((size_t)N_NODES * DM);
  int* nzL = (int*)alloc(N_NODES);
  int* nzR = (int*)alloc(N_NODES);
  float* left  = (float*)alloc((size_t)N_NODES * DM);
  float* right = (float*)alloc((size_t)N_NODES * DM);
  float* feats = (float*)alloc((size_t)N_NODES * (REP + 2 * DM));
  float* c1    = (float*)alloc((size_t)N_NODES * 512);

  const int T = 256;

  // ---- GAT layer 1 (8 heads x 64)
  gemm(x_param, F_IN, W1, HEADS * HID, xw1, HEADS * HID, N_NODES, HEADS * HID, F_IN, nullptr, 0);
  attn_dots<<<cdiv((long long)N_NODES * HEADS, T), T, 0, stream>>>(xw1, att_src1, att_dst1, asrc1, adst1, HEADS, HID);
  zero(mkey1, (size_t)N_NODES * HEADS);
  edge_scores<<<cdiv((long long)EDGES_TOT * HEADS, T), T, 0, stream>>>(asrc1, adst1, edge_index, esc1, mkey1, HEADS);
  zero(den1, (size_t)N_NODES * HEADS);
  edge_exp<<<cdiv((long long)EDGES_TOT * HEADS, T), T, 0, stream>>>(esc1, mkey1, den1, edge_index, HEADS);
  zero(agg1, (size_t)N_NODES * HEADS * HID);
  edge_agg<<<cdiv((long long)EDGES_TOT * HEADS * HID, T), T, 0, stream>>>(xw1, esc1, den1, edge_index, agg1, HEADS, HID);
  bias_act<<<cdiv((long long)N_NODES * HEADS * HID, T), T, 0, stream>>>(h1, agg1, b1, (long long)N_NODES * HEADS * HID, HEADS * HID, 3);

  // ---- GAT layer 2 (1 head x 512, mean == identity)
  gemm(h1, HEADS * HID, W2, REP, xw2, REP, N_NODES, REP, HEADS * HID, nullptr, 0);
  attn_dots<<<cdiv((long long)N_NODES, T), T, 0, stream>>>(xw2, att_src2, att_dst2, asrc2, adst2, 1, REP);
  zero(mkey2, N_NODES);
  edge_scores<<<cdiv((long long)EDGES_TOT, T), T, 0, stream>>>(asrc2, adst2, edge_index, esc2, mkey2, 1);
  zero(den2, N_NODES);
  edge_exp<<<cdiv((long long)EDGES_TOT, T), T, 0, stream>>>(esc2, mkey2, den2, edge_index, 1);
  zero(agg2, (size_t)N_NODES * REP);
  edge_agg<<<cdiv((long long)EDGES_TOT * REP, T), T, 0, stream>>>(xw2, esc2, den2, edge_index, agg2, 1, REP);
  bias_act<<<cdiv((long long)N_NODES * REP, T), T, 0, stream>>>(h2, agg2, b2, (long long)N_NODES * REP, REP, 0);

  // ---- Mamba encoder
  zero(counts, E_EDGES);
  seq_counts<<<cdiv(NSYN, T), T, 0, stream>>>(synapse_index, counts);
  seq_starts<<<1, 1, 0, stream>>>(counts, starts);
  zero(padded, (size_t)SEQ_TOT * 6);
  seq_scatter<<<cdiv(NSYN, T), T, 0, stream>>>(synapse, synapse_index, starts, padded);
  gemm(padded, 6, W_in, DM, u, DM, SEQ_TOT, DM, 6, b_in, 0);
  gemm(u, DM, W_inproj, 2 * DI, xz, 2 * DI, SEQ_TOT, 2 * DI, DM, nullptr, 0);
  dwconv_silu<<<cdiv((long long)SEQ_TOT * DI, T), T, 0, stream>>>(xz, conv_w, conv_b, xc);
  gemm(xc, DI, W_xproj, DTR + 2 * DSTATE, xdb, DTR + 2 * DSTATE, SEQ_TOT, DTR + 2 * DSTATE, DI, nullptr, 0);
  gemm(xdb, DTR + 2 * DSTATE, W_dt, DI, dtb, DI, SEQ_TOT, DI, DTR, b_dt, 2 /*softplus*/);
  ssm_scan<<<cdiv((long long)E_EDGES * DI, T), T, 0, stream>>>(dtb, xdb, xc, A_log, Dvec, yb);
  gate_silu<<<cdiv((long long)SEQ_TOT * DI, T), T, 0, stream>>>(yb, xz);
  gemm(yb, DI, W_outproj, DM, outseq, DM, SEQ_TOT, DM, DI, nullptr, 0);
  pool_mean<<<cdiv((long long)E_EDGES * DM, T), T, 0, stream>>>(outseq, counts, pooled);
  gemm(pooled, DM, W_lout, DM, xpoint, DM, E_EDGES, DM, DM, b_lout, 0);

  // ---- per-endpoint scatter-max of edge features
  zero(keyL, (size_t)N_NODES * DM); zero(nzL, N_NODES);
  zero(keyR, (size_t)N_NODES * DM); zero(nzR, N_NODES);
  seg_scatter<<<cdiv((long long)E_EDGES * DM, T), T, 0, stream>>>(xpoint, edge_index,           keyL, nzL);
  seg_scatter<<<cdiv((long long)E_EDGES * DM, T), T, 0, stream>>>(xpoint, edge_index + E_EDGES, keyR, nzR);
  seg_final<<<cdiv((long long)N_NODES * DM, T), T, 0, stream>>>(keyL, nzL, left);
  seg_final<<<cdiv((long long)N_NODES * DM, T), T, 0, stream>>>(keyR, nzR, right);

  // ---- classifier
  build_feats<<<cdiv((long long)N_NODES * (REP + 2 * DM), T), T, 0, stream>>>(h2, left, right, feats);
  gemm(feats, REP + 2 * DM, Wc1, 512, c1, 512, N_NODES, 512, REP + 2 * DM, bc1, 1 /*relu*/);
  gemm(c1, 512, Wc2, NCLS, (float*)d_out, NCLS, N_NODES, NCLS, 512, bc2, 0);
}
